// NLSALayer_10591389352526
// MI455X (gfx1250) — matrix operand order repaired and checked by reading, not compiled, not run on hardware
//
#include <hip/hip_runtime.h>
#include <hip/hip_bf16.h>

// ---------------------------------------------------------------------------
// NLSA layer for MI455X (gfx1250, wave32, WMMA).
// conv_match(3^3,64->16) -> conv_assembly(1x1,64->64, WMMA GEMM) -> LSH hash
//   -> stable counting sort (codes in [0,384)) -> gather -> transposed
//   flash-attention with v_wmma_f32_16x16x32_bf16 -> unsort+combine -> conv.
// All WMMA operand LDS reads are 16B-aligned contiguous chunks (ds_load_b128),
// LDS strides chosen bank-conflict-free over 16 lanes.
// ---------------------------------------------------------------------------

typedef __bf16 bf16_t;
typedef __attribute__((ext_vector_type(16))) __bf16 v16bf;
typedef __attribute__((ext_vector_type(8)))  __bf16 v8bf;
typedef __attribute__((ext_vector_type(8)))  float  v8f;
typedef __attribute__((ext_vector_type(4)))  float  v4f;

constexpr int B_   = 2;
constexpr int C_   = 64;
constexpr int CR   = 16;
constexpr int L_   = 13824;        // 24^3
constexpr int NH   = 4;
constexpr int CHK  = 144;
constexpr int NCH  = 96;           // L / CHUNK
constexpr int NBUK = 96;           // hash buckets per round
constexpr int BINS = NH * NBUK;    // 384 total bucket codes
constexpr int N4   = NH * L_;      // 55296 occurrences per batch
constexpr int NBLK = N4 / 256;     // 216 sort blocks per batch

__device__ __forceinline__ bf16_t f2bf(float f) {
  unsigned u = __builtin_bit_cast(unsigned, f);
  unsigned r = u + 0x7FFFu + ((u >> 16) & 1u);   // round-to-nearest-even
  return __builtin_bit_cast(bf16_t, (unsigned short)(r >> 16));
}

// ---------------------------------------------------------------------------
// conv3d 3x3x3, 64 -> 16, bias, token-major output xe[b*L+l][16]
// ---------------------------------------------------------------------------
__global__ __launch_bounds__(256) void conv_match_kernel(
    const float* __restrict__ x, const float* __restrict__ w,
    const float* __restrict__ bias, float* __restrict__ xe)
{
  extern __shared__ float wl[];               // [tap][ci][co] 27*64*16
  for (int i = threadIdx.x; i < 27*64*16; i += 256) {
    int co = i & 15, ci = (i >> 4) & 63, tap = i >> 10;
    wl[i] = w[(co*64 + ci)*27 + tap];
  }
  __syncthreads();
  int g = blockIdx.x * 256 + threadIdx.x;     // (b, voxel)
  int b = g / L_, v = g - b * L_;
  int z = v / 576, rem = v - z * 576, yy = rem / 24, xx = rem - yy * 24;
  float acc[16];
#pragma unroll
  for (int co = 0; co < 16; ++co) acc[co] = 0.f;
  const float* xb = x + (size_t)b * C_ * L_;
  for (int kd = 0; kd < 3; ++kd) {
    int zz = z + kd - 1; if (zz < 0 || zz >= 24) continue;
    for (int kh = 0; kh < 3; ++kh) {
      int yi = yy + kh - 1; if (yi < 0 || yi >= 24) continue;
      for (int kw = 0; kw < 3; ++kw) {
        int xi = xx + kw - 1; if (xi < 0 || xi >= 24) continue;
        int tap = (kd*3 + kh)*3 + kw;
        const float* xp = xb + zz*576 + yi*24 + xi;
        const float* wt = &wl[tap * 1024];
        for (int ci = 0; ci < 64; ++ci) {
          float xv = xp[(size_t)ci * L_];
          const float* wc = wt + ci*16;
#pragma unroll
          for (int co = 0; co < 16; ++co) acc[co] = fmaf(xv, wc[co], acc[co]);
        }
      }
    }
  }
#pragma unroll
  for (int co = 0; co < 16; ++co) xe[(size_t)g * 16 + co] = acc[co] + bias[co];
}

// ---------------------------------------------------------------------------
// conv3d 1x1x1, 64 -> 64 as a WMMA GEMM: ye[token][co] = x[token][:] * W + b
// Block = 144 tokens, 9 waves; wave owns a 16-token tile; 8 wmma per wave.
// W kept [cout][cin] in LDS so B-fragment reads are contiguous along cin.
// ---------------------------------------------------------------------------
constexpr int XSTR = 72;   // bf16 [144 tokens][XSTR]    (144B rows, 16B aligned)
constexpr int WSTR = 72;   // bf16 [64 cout][WSTR]
constexpr int ASM_LDS = (144*XSTR + 64*WSTR) * (int)sizeof(bf16_t) + 64*4;

__global__ __launch_bounds__(288) void conv_assembly_kernel(
    const float* __restrict__ x, const float* __restrict__ w,
    const float* __restrict__ bias, float* __restrict__ ye)
{
  extern __shared__ char smem[];
  bf16_t* Xl = (bf16_t*)smem;                 // [token][cin]
  bf16_t* Wl = Xl + 144*XSTR;                 // [cout][cin]
  float*  bl = (float*)(Wl + 64*WSTR);
  int tid = threadIdx.x;
  int blk = blockIdx.x;                       // 192 blocks (96 per batch)
  int b = blk / 96;
  int l0 = (blk - b * 96) * 144;
  for (int e = tid; e < 64*144; e += 288) {
    int ci = e / 144, t = e - ci*144;
    Xl[t*XSTR + ci] = f2bf(x[((size_t)b*C_ + ci)*L_ + l0 + t]);
  }
  for (int e = tid; e < 64*64; e += 288) {
    int co = e >> 6, ci = e & 63;
    Wl[co*WSTR + ci] = f2bf(w[co*64 + ci]);
  }
  if (tid < 64) bl[tid] = bias[tid];
  __syncthreads();

  int wave = tid >> 5, lane = tid & 31;
  int lrow = lane & 15, hi = lane >> 4;
  int rowbase = wave * 16;
  int kbA = hi ? 8 : 0, kbB = hi ? 16 : 0;

  v8f O0 = {}, O1 = {}, O2 = {}, O3 = {};
#pragma unroll
  for (int ks = 0; ks < 64; ks += 32) {
    v16bf a;
    const bf16_t* ap = &Xl[(rowbase + lrow)*XSTR + ks];
#pragma unroll
    for (int i = 0; i < 8; ++i) { a[i] = ap[kbB + i]; a[8+i] = ap[kbB + 8 + i]; }
    // B[k=ci][n=co]: contiguous along ci per output-channel row
    v16bf b0, b1, b2, b3;
    const bf16_t* w0 = &Wl[(0*16 + lrow)*WSTR + ks + kbB];
    const bf16_t* w1 = &Wl[(1*16 + lrow)*WSTR + ks + kbB];
    const bf16_t* w2 = &Wl[(2*16 + lrow)*WSTR + ks + kbB];
    const bf16_t* w3 = &Wl[(3*16 + lrow)*WSTR + ks + kbB];
#pragma unroll
    for (int i = 0; i < 16; ++i) { b0[i] = w0[i]; b1[i] = w1[i]; b2[i] = w2[i]; b3[i] = w3[i]; }
    O0 = __builtin_amdgcn_wmma_f32_16x16x32_bf16(false, a, false, b0, (short)0, O0, false, false);
    O1 = __builtin_amdgcn_wmma_f32_16x16x32_bf16(false, a, false, b1, (short)0, O1, false, false);
    O2 = __builtin_amdgcn_wmma_f32_16x16x32_bf16(false, a, false, b2, (short)0, O2, false, false);
    O3 = __builtin_amdgcn_wmma_f32_16x16x32_bf16(false, a, false, b3, (short)0, O3, false, false);
  }
#pragma unroll
  for (int r = 0; r < 8; ++r) {
    int tok = rowbase + r + 8*hi;
    float* op = ye + ((size_t)b*L_ + l0 + tok)*C_ + lrow;
    op[0]  = O0[r] + bl[lrow];
    op[16] = O1[r] + bl[16 + lrow];
    op[32] = O2[r] + bl[32 + lrow];
    op[48] = O3[r] + bl[48 + lrow];
  }
}

// ---------------------------------------------------------------------------
// LSH hashing: codes[b][h*L+l] = argmax([rv, -rv]) + h*96  (single pass)
// ---------------------------------------------------------------------------
__global__ __launch_bounds__(256) void hash_kernel(
    const float* __restrict__ xe, const float* __restrict__ rot,
    int* __restrict__ codes)
{
  __shared__ float rl[CR * NH * 48];
  for (int i = threadIdx.x; i < CR*NH*48; i += 256) rl[i] = rot[i];
  __syncthreads();
  int g = blockIdx.x * 256 + threadIdx.x;
  int b = g / N4; int r = g - b * N4;
  int h = r / L_; int l = r - h * L_;
  const float* xp = xe + (size_t)(b * L_ + l) * CR;
  float xv[CR];
#pragma unroll
  for (int c = 0; c < CR; ++c) xv[c] = xp[c];
  float bestP = -3.0e38f, bestN = -3.0e38f;
  int idxP = 0, idxN = 0;
  for (int i = 0; i < 48; ++i) {
    float d = 0.f;
#pragma unroll
    for (int c = 0; c < CR; ++c) d = fmaf(xv[c], rl[(c*NH + h)*48 + i], d);
    if (d > bestP)  { bestP = d;  idxP = i; }
    if (-d > bestN) { bestN = -d; idxN = i; }
  }
  // all '+' indices (0..47) precede all '-' indices (48..95): ties go to '+'
  int bi = (bestP >= bestN) ? idxP : 48 + idxN;
  codes[b * N4 + h * L_ + l] = bi + h * NBUK;
}

// ---------------------------------------------------------------------------
// Stable counting sort of codes (384 bins) -> argsort + inverse permutation.
// ---------------------------------------------------------------------------
__global__ __launch_bounds__(256) void sort_hist_kernel(
    const int* __restrict__ codes, int* __restrict__ bh)
{
  __shared__ int h[BINS];
  int b = blockIdx.x / NBLK, blk = blockIdx.x - b * NBLK;
  for (int i = threadIdx.x; i < BINS; i += 256) h[i] = 0;
  __syncthreads();
  int code = codes[b * N4 + blk * 256 + threadIdx.x];
  atomicAdd(&h[code], 1);
  __syncthreads();
  for (int i = threadIdx.x; i < BINS; i += 256)
    bh[(b * BINS + i) * NBLK + blk] = h[i];
}

__global__ __launch_bounds__(256) void sort_scan_kernel(
    const int* __restrict__ bh, int* __restrict__ boff)
{
  __shared__ int tot[BINS];
  __shared__ int binStart[BINS];
  int b = blockIdx.x;
  for (int bin = threadIdx.x; bin < BINS; bin += 256) {
    int s = 0;
    const int* p = bh + (size_t)(b * BINS + bin) * NBLK;
    for (int k = 0; k < NBLK; ++k) s += p[k];
    tot[bin] = s;
  }
  __syncthreads();
  if (threadIdx.x == 0) {
    int acc = 0;
    for (int i = 0; i < BINS; ++i) { binStart[i] = acc; acc += tot[i]; }
  }
  __syncthreads();
  for (int bin = threadIdx.x; bin < BINS; bin += 256) {
    int acc = binStart[bin];
    const int* p = bh + (size_t)(b * BINS + bin) * NBLK;
    int* q = boff + (size_t)(b * BINS + bin) * NBLK;
    for (int k = 0; k < NBLK; ++k) { q[k] = acc; acc += p[k]; }
  }
}

__global__ __launch_bounds__(256) void sort_place_kernel(
    const int* __restrict__ codes, const int* __restrict__ boff,
    int* __restrict__ sidx, int* __restrict__ undo_)
{
  __shared__ int lc[256];
  int b = blockIdx.x / NBLK, blk = blockIdx.x - b * NBLK;
  int gi = blk * 256 + threadIdx.x;
  int code = codes[b * N4 + gi];
  lc[threadIdx.x] = code;
  __syncthreads();
  int rank = 0;
  for (int j = 0; j < (int)threadIdx.x; ++j) rank += (lc[j] == code) ? 1 : 0;
  int pos = boff[(b * BINS + code) * NBLK + blk] + rank;
  sidx[b * N4 + pos] = gi;
  undo_[b * N4 + gi] = pos;
}

// ---------------------------------------------------------------------------
// Gather sorted sequences: xs f32 [B][4L][16], ys bf16 [B][4L][64]
// ---------------------------------------------------------------------------
__global__ __launch_bounds__(256) void gather_kernel(
    const int* __restrict__ sidx, const float* __restrict__ xe,
    const float* __restrict__ ye, float* __restrict__ xs,
    bf16_t* __restrict__ ysb)
{
  int b = blockIdx.x / NBLK;
  int s = (blockIdx.x - b * NBLK) * 256 + threadIdx.x;
  int idx = sidx[b * N4 + s];
  int l = idx % L_;
  const float* xp = xe + (size_t)(b * L_ + l) * CR;
  float* xo = xs + (size_t)(b * N4 + s) * CR;
#pragma unroll
  for (int c = 0; c < CR; ++c) xo[c] = xp[c];
  const float* yp = ye + (size_t)(b * L_ + l) * C_;
  bf16_t* yo = ysb + (size_t)(b * N4 + s) * C_;
#pragma unroll
  for (int c = 0; c < C_; ++c) yo[c] = f2bf(yp[c]);
}

// ---------------------------------------------------------------------------
// Transposed flash attention. One block (9 waves) per (b,h,chunk); wave w owns
// queries [16w,16w+16). S^T = K x Q^T so the query is the lane index: softmax
// stats are per-lane scalars needing only one xor-16 shuffle per statistic.
// O^T += V^T x P^T; P staged [query][key] via two 16B LDS stores per lane.
// ---------------------------------------------------------------------------
constexpr int QSTR  = 40;   // bf16 strides: mult of 8 (16B aligned chunks),
constexpr int KSTR  = 40;   // bank step 20 -> conflict-free over 16 lanes
constexpr int YSTR  = 456;  // bank step 36 -> conflict-free
constexpr int PSTR  = 40;
constexpr int OTSTR = 68;   // f32 stride: bank step 4 -> conflict-free
constexpr int ATTN_LDS =
    (CHK*QSTR + 448*KSTR + C_*YSTR + 9*16*PSTR) * (int)sizeof(bf16_t)
    + 9*16*OTSTR * (int)sizeof(float);

__global__ __launch_bounds__(288) void attn_kernel(
    const float* __restrict__ xs, const bf16_t* __restrict__ ysb,
    float* __restrict__ retb, float* __restrict__ bsb)
{
  extern __shared__ char smem[];
  bf16_t* Qs  = (bf16_t*)smem;               // [144][QSTR] cols 0..15 data, 16..31 zero
  bf16_t* Ks  = Qs + CHK * QSTR;             // [448][KSTR] normalized keys
  bf16_t* YsT = Ks + 448 * KSTR;             // [64][YSTR]  values channel-major
  bf16_t* Pt  = YsT + C_ * YSTR;             // [9][16 q][PSTR]
  float*  Ot  = (float*)(Pt + 9*16*PSTR);    // [9][16 q][OTSTR]

  int bid = blockIdx.x;
  int k = bid % NCH; int h = (bid / NCH) & 3; int b = bid / (NCH * NH);
  int base = (b * NH + h) * L_;
  int tid = threadIdx.x;
  int km1 = (k + NCH - 1) % NCH, kp1 = (k + 1) % NCH;

  for (int r = tid; r < CHK; r += 288) {            // queries (unnormalized)
    const float* src = xs + (size_t)(base + k * CHK + r) * CR;
    bf16_t* q = &Qs[r * QSTR];
#pragma unroll
    for (int c = 0; c < CR; ++c) q[c] = f2bf(src[c]);
#pragma unroll
    for (int c = CR; c < 32; ++c) q[c] = f2bf(0.f);
  }
  for (int r = tid; r < 448; r += 288) {            // keys, L2-normalized
    bf16_t* kd = &Ks[r * KSTR];
    if (r < 3 * CHK) {
      int cat = r / CHK, rr = r - cat * CHK;
      int kc = (cat == 0) ? k : ((cat == 1) ? km1 : kp1);
      const float* src = xs + (size_t)(base + kc * CHK + rr) * CR;
      float v[CR]; float ss = 0.f;
#pragma unroll
      for (int c = 0; c < CR; ++c) { v[c] = src[c]; ss = fmaf(v[c], v[c], ss); }
      float inv = 1.0f / fmaxf(sqrtf(ss), 5e-5f);
#pragma unroll
      for (int c = 0; c < CR; ++c) kd[c] = f2bf(v[c] * inv);
#pragma unroll
      for (int c = CR; c < 32; ++c) kd[c] = f2bf(0.f);
    } else {
#pragma unroll
      for (int c = 0; c < 32; ++c) kd[c] = f2bf(0.f);
    }
  }
  for (int e = tid; e < 448 * C_; e += 288) {       // values -> channel-major
    int r = e >> 6, c = e & 63;
    bf16_t v = f2bf(0.f);
    if (r < 3 * CHK) {
      int cat = r / CHK, rr = r - cat * CHK;
      int kc = (cat == 0) ? k : ((cat == 1) ? km1 : kp1);
      v = ysb[(size_t)(base + kc * CHK + rr) * C_ + c];
    }
    YsT[c * YSTR + r] = v;
  }
  __syncthreads();

  int wave = tid >> 5, lane = tid & 31;
  int lrow = lane & 15, hi = lane >> 4;
  int rowbase = wave * 16;
  int kbA = hi ? 8 : 0, kbB = hi ? 16 : 0;

  // Q as B-fragment (channels x queries): b[i] = Q[q=rowbase+lrow][c=kbB+i]
  v16bf qb;
  {
    const bf16_t* qp = &Qs[(rowbase + lrow) * QSTR + kbB];
#pragma unroll
    for (int i = 0; i < 16; ++i) qb[i] = qp[i];
  }

  v8f O0 = {}, O1 = {}, O2 = {}, O3 = {};
  float m = -1.0e30f, lsum = 0.f;
  bf16_t* Pw = &Pt[wave * 16 * PSTR];

#pragma unroll 2
  for (int jj = 0; jj < 14; ++jj) {
    int kb0 = jj * 32;
    const bool valid1 = (jj < 13);

    // keys as A-fragments: a[i] = K[key = kb0(+16)+lrow][c]
    v16bf ka0, ka1;
    {
      const bf16_t* kp0 = &Ks[(kb0 + lrow) * KSTR];
      const bf16_t* kp1 = &Ks[(kb0 + 16 + lrow) * KSTR];
#pragma unroll
      for (int i = 0; i < 8; ++i) {
        ka0[i] = kp0[kbB + i]; ka0[8+i] = kp0[kbB + 8 + i];
        ka1[i] = kp1[kbB + i]; ka1[8+i] = kp1[kbB + 8 + i];
      }
    }
    v8f s0 = {}, s1 = {};
    s0 = __builtin_amdgcn_wmma_f32_16x16x32_bf16(false, ka0, false, qb, (short)0, s0, false, false);
    s1 = __builtin_amdgcn_wmma_f32_16x16x32_bf16(false, ka1, false, qb, (short)0, s1, false, false);

    // per-lane (= per-query) online softmax; 16 key-scores live in registers
    float mx = s0[0];
#pragma unroll
    for (int r = 1; r < 8; ++r) mx = fmaxf(mx, s0[r]);
    if (valid1) {
#pragma unroll
      for (int r = 0; r < 8; ++r) mx = fmaxf(mx, s1[r]);
    }
    mx = fmaxf(mx, __shfl_xor(mx, 16, 32));
    float mnew = fmaxf(m, mx);
    float sc = __expf(m - mnew);
    float e0[8], e1[8];
    float rs = 0.f;
#pragma unroll
    for (int r = 0; r < 8; ++r) { e0[r] = __expf(s0[r] - mnew); rs += e0[r]; }
#pragma unroll
    for (int r = 0; r < 8; ++r) e1[r] = 0.f;
    if (valid1) {
#pragma unroll
      for (int r = 0; r < 8; ++r) { e1[r] = __expf(s1[r] - mnew); rs += e1[r]; }
    }
    rs += __shfl_xor(rs, 16, 32);
    lsum = lsum * sc + rs;
    m = mnew;
    O0 *= sc; O1 *= sc; O2 *= sc; O3 *= sc;

    // stage P^T as [q][key_local] with two 16B stores per lane
    v8bf pe0, pe1;
#pragma unroll
    for (int r = 0; r < 8; ++r) { pe0[r] = f2bf(e0[r]); pe1[r] = f2bf(e1[r]); }
    *(v8bf*)&Pw[lrow * PSTR + 8*hi]      = pe0;
    *(v8bf*)&Pw[lrow * PSTR + 16 + 8*hi] = pe1;
    asm volatile("s_wait_dscnt 0" ::: "memory");
    v16bf pb;
    {
      const bf16_t* pp = &Pw[lrow * PSTR + kbB];
#pragma unroll
      for (int i = 0; i < 16; ++i) pb[i] = pp[i];
    }

    // O^T += V^T x P^T (K = 32 keys fully used)
    {
      const bf16_t* v0 = &YsT[(0*16 + lrow) * YSTR + kb0];
      const bf16_t* v1 = &YsT[(1*16 + lrow) * YSTR + kb0];
      const bf16_t* v2 = &YsT[(2*16 + lrow) * YSTR + kb0];
      const bf16_t* v3 = &YsT[(3*16 + lrow) * YSTR + kb0];
      v16bf a0, a1, a2, a3;
#pragma unroll
      for (int i = 0; i < 8; ++i) {
        a0[i] = v0[kbB + i]; a0[8+i] = v0[kbB + 8 + i];
        a1[i] = v1[kbB + i]; a1[8+i] = v1[kbB + 8 + i];
        a2[i] = v2[kbB + i]; a2[8+i] = v2[kbB + 8 + i];
        a3[i] = v3[kbB + i]; a3[8+i] = v3[kbB + 8 + i];
      }
      O0 = __builtin_amdgcn_wmma_f32_16x16x32_bf16(false, a0, false, pb, (short)0, O0, false, false);
      O1 = __builtin_amdgcn_wmma_f32_16x16x32_bf16(false, a1, false, pb, (short)0, O1, false, false);
      O2 = __builtin_amdgcn_wmma_f32_16x16x32_bf16(false, a2, false, pb, (short)0, O2, false, false);
      O3 = __builtin_amdgcn_wmma_f32_16x16x32_bf16(false, a3, false, pb, (short)0, O3, false, false);
    }
  }

  // normalize, transpose via LDS (v4f stores), coalesced v4f global stores
  float inv = 1.0f / lsum;
  float* Ow = &Ot[wave * 16 * OTSTR];
  {
    float* d0 = &Ow[lrow * OTSTR + 8*hi];
    *(v4f*)(d0 + 0)  = (v4f){O0[0]*inv, O0[1]*inv, O0[2]*inv, O0[3]*inv};
    *(v4f*)(d0 + 4)  = (v4f){O0[4]*inv, O0[5]*inv, O0[6]*inv, O0[7]*inv};
    float* d1 = &Ow[lrow * OTSTR + 16 + 8*hi];
    *(v4f*)(d1 + 0)  = (v4f){O1[0]*inv, O1[1]*inv, O1[2]*inv, O1[3]*inv};
    *(v4f*)(d1 + 4)  = (v4f){O1[4]*inv, O1[5]*inv, O1[6]*inv, O1[7]*inv};
    float* d2 = &Ow[lrow * OTSTR + 32 + 8*hi];
    *(v4f*)(d2 + 0)  = (v4f){O2[0]*inv, O2[1]*inv, O2[2]*inv, O2[3]*inv};
    *(v4f*)(d2 + 4)  = (v4f){O2[4]*inv, O2[5]*inv, O2[6]*inv, O2[7]*inv};
    float* d3 = &Ow[lrow * OTSTR + 48 + 8*hi];
    *(v4f*)(d3 + 0)  = (v4f){O3[0]*inv, O3[1]*inv, O3[2]*inv, O3[3]*inv};
    *(v4f*)(d3 + 4)  = (v4f){O3[4]*inv, O3[5]*inv, O3[6]*inv, O3[7]*inv};
  }
  asm volatile("s_wait_dscnt 0" ::: "memory");
  for (int e = lane; e < 16 * 16; e += 32) {        // 256 x v4f
    int q = e >> 4, c4 = (e & 15) << 2;
    v4f v = *(const v4f*)&Ow[q * OTSTR + c4];
    *(v4f*)&retb[(size_t)(base + k * CHK + rowbase + q) * C_ + c4] = v;
  }
  if (hi == 0)
    bsb[base + k * CHK + rowbase + lrow] = m + __logf(lsum);
}

// ---------------------------------------------------------------------------
// Unsort + softmax over hash rounds + residual: y[b][c][l]
// ---------------------------------------------------------------------------
__global__ __launch_bounds__(256) void combine_kernel(
    const int* __restrict__ undo_, const float* __restrict__ bsb,
    const float* __restrict__ retb, const float* __restrict__ x,
    float* __restrict__ y)
{
  int g = blockIdx.x * 256 + threadIdx.x;
  int b = g / L_, l = g - b * L_;
  int pos[NH]; float bsv[NH];
#pragma unroll
  for (int h = 0; h < NH; ++h) {
    pos[h] = undo_[b * N4 + h * L_ + l];
    bsv[h] = bsb[b * N4 + pos[h]];
  }
  float mx = bsv[0];
#pragma unroll
  for (int h = 1; h < NH; ++h) mx = fmaxf(mx, bsv[h]);
  float wgt[NH]; float sum = 0.f;
#pragma unroll
  for (int h = 0; h < NH; ++h) { wgt[h] = __expf(bsv[h] - mx); sum += wgt[h]; }
  float inv = 1.0f / sum;
#pragma unroll
  for (int h = 0; h < NH; ++h) wgt[h] *= inv;
  for (int c = 0; c < C_; ++c) {
    float o = 0.f;
#pragma unroll
    for (int h = 0; h < NH; ++h)
      o = fmaf(wgt[h], retb[((size_t)b * N4 + pos[h]) * C_ + c], o);
    size_t oi = ((size_t)b * C_ + c) * L_ + l;
    y[oi] = o + x[oi];
  }
}

// ---------------------------------------------------------------------------
// Final conv3d 3x3x3 64->64 on relu(y), +bias, +y residual -> d_out
// ---------------------------------------------------------------------------
__global__ __launch_bounds__(256) void conv_final_kernel(
    const float* __restrict__ y, const float* __restrict__ w,
    const float* __restrict__ bias, float* __restrict__ out)
{
  extern __shared__ float wl[];               // [tap][ci][co16]
  int coT = blockIdx.y;
  for (int i = threadIdx.x; i < 27*64*16; i += 256) {
    int col = i & 15, ci = (i >> 4) & 63, tap = i >> 10;
    wl[i] = w[((coT*16 + col)*64 + ci)*27 + tap];
  }
  __syncthreads();
  int g = blockIdx.x * 256 + threadIdx.x;
  int b = g / L_, v = g - b * L_;
  int z = v / 576, rem = v - z * 576, yy = rem / 24, xx = rem - yy * 24;
  float acc[16];
#pragma unroll
  for (int co = 0; co < 16; ++co) acc[co] = 0.f;
  const float* yb = y + (size_t)b * C_ * L_;
  for (int kd = 0; kd < 3; ++kd) {
    int zz = z + kd - 1; if (zz < 0 || zz >= 24) continue;
    for (int kh = 0; kh < 3; ++kh) {
      int yi = yy + kh - 1; if (yi < 0 || yi >= 24) continue;
      for (int kw = 0; kw < 3; ++kw) {
        int xi = xx + kw - 1; if (xi < 0 || xi >= 24) continue;
        int tap = (kd*3 + kh)*3 + kw;
        const float* yp = yb + zz*576 + yi*24 + xi;
        const float* wt = &wl[tap * 1024];
        for (int ci = 0; ci < 64; ++ci) {
          float xv = fmaxf(yp[(size_t)ci * L_], 0.f);   // relu
          const float* wc = wt + ci*16;
#pragma unroll
          for (int co = 0; co < 16; ++co) acc[co] = fmaf(xv, wc[co], acc[co]);
        }
      }
    }
  }
#pragma unroll
  for (int co = 0; co < 16; ++co) {
    int cog = coT*16 + co;
    size_t oi = ((size_t)b * C_ + cog) * L_ + v;
    out[oi] = y[oi] + acc[co] + bias[cog];
  }
}

// ---------------------------------------------------------------------------
extern "C" void kernel_launch(void* const* d_in, const int* in_sizes, int n_in,
                              void* d_out, int out_size, void* d_ws, size_t ws_size,
                              hipStream_t stream) {
  const float* x      = (const float*)d_in[0];
  const float* wmatch = (const float*)d_in[1];
  const float* bmatch = (const float*)d_in[2];
  const float* wasm   = (const float*)d_in[3];
  const float* basm   = (const float*)d_in[4];
  const float* rot    = (const float*)d_in[5];
  const float* wconv  = (const float*)d_in[6];
  const float* bconv  = (const float*)d_in[7];
  float* out = (float*)d_out;

  size_t cur = 0;
  auto alloc = [&](size_t bytes) -> void* {
    void* p = (char*)d_ws + cur;
    cur += (bytes + 255) & ~(size_t)255;
    return p;
  };
  float*  xe    = (float*) alloc((size_t)B_ * L_ * CR * 4);
  float*  ye    = (float*) alloc((size_t)B_ * L_ * C_ * 4);
  int*    codes = (int*)   alloc((size_t)B_ * N4 * 4);
  int*    bh    = (int*)   alloc((size_t)B_ * BINS * NBLK * 4);
  int*    boff  = (int*)   alloc((size_t)B_ * BINS * NBLK * 4);
  int*    sidx  = (int*)   alloc((size_t)B_ * N4 * 4);
  int*    undo_ = (int*)   alloc((size_t)B_ * N4 * 4);
  float*  xs    = (float*) alloc((size_t)B_ * N4 * CR * 4);
  bf16_t* ysb   = (bf16_t*)alloc((size_t)B_ * N4 * C_ * 2);
  float*  retb  = (float*) alloc((size_t)B_ * N4 * C_ * 4);
  float*  bsb   = (float*) alloc((size_t)B_ * N4 * 4);
  float*  yb    = (float*) alloc((size_t)B_ * C_ * L_ * 4);
  (void)ws_size; (void)in_sizes; (void)n_in; (void)out_size;

  const int convLds = 27 * 64 * 16 * (int)sizeof(float);

  conv_match_kernel   <<<B_ * L_ / 256, 256, convLds, stream>>>(x, wmatch, bmatch, xe);
  conv_assembly_kernel<<<B_ * L_ / CHK, 288, ASM_LDS, stream>>>(x, wasm, basm, ye);
  hash_kernel         <<<B_ * N4 / 256, 256, 0, stream>>>(xe, rot, codes);
  sort_hist_kernel    <<<B_ * NBLK, 256, 0, stream>>>(codes, bh);
  sort_scan_kernel    <<<B_, 256, 0, stream>>>(bh, boff);
  sort_place_kernel   <<<B_ * NBLK, 256, 0, stream>>>(codes, boff, sidx, undo_);
  gather_kernel       <<<B_ * NBLK, 256, 0, stream>>>(sidx, xe, ye, xs, ysb);
  attn_kernel         <<<B_ * NH * NCH, 288, ATTN_LDS, stream>>>(xs, ysb, retb, bsb);
  combine_kernel      <<<B_ * L_ / 256, 256, 0, stream>>>(undo_, bsb, retb, x, yb);
  conv_final_kernel   <<<dim3(B_ * L_ / 256, 4), 256, convLds, stream>>>(yb, wconv, bconv, out);
}
